// SelfAttention_29437705847488
// MI455X (gfx1250) — matrix operand order repaired and checked by reading, not compiled, not run on hardware
//
#include <hip/hip_runtime.h>
#include <hip/hip_bf16.h>

// ---------------------------------------------------------------------------
// MI455X (gfx1250) self-attention, bf16 WMMA pipeline, software-pipelined.
//
// Roofline: ~137 GFLOP vs ~150 MB HBM traffic -> compute-bound; everything
// rides v_wmma_f32_16x16x32_bf16 (K=32/instr, 8x the K-rate of the f32 WMMA).
// v2 changes vs v1 (driven by disasm showing `s_wait_loadcnt 0x0` before
// every WMMA = fully serialized load->MMA):
//   * GEMMs: 32x64 register tile/wave (8 acc), double-buffered A/B fragments
//     so next k-step's loads are in flight during current WMMAs.
//   * Attention: V-frags + next-K-frags issued before the S WMMAs/softmax so
//     the VALU/exp chunk and LDS P-relayout hide their latency.
// ---------------------------------------------------------------------------

typedef __bf16 bf16;
typedef __attribute__((ext_vector_type(16))) __bf16 v16bf;
typedef __attribute__((ext_vector_type(8)))  __bf16 v8bf;
typedef __attribute__((ext_vector_type(8)))  float  v8f;

#define EMBED 1024
#define NHEAD 16
#define HDIM  64
#define SEQ   2048
#define BATCH 4
#define TDIM  (3 * EMBED)          // 3072
#define MROWS (BATCH * SEQ)        // 8192

// ---- WMMA fragment helpers (layouts per CDNA5 ISA 7.12.2, wave32) ----------

// A-matrix 16x32 bf16 tile at (m0, k0) of row-major A[*, lda].
// lane m = lane&15 holds row m; kh = lane>>4 selects K-half.
__device__ __forceinline__ v16bf load_frag_a(const bf16* __restrict__ A, int lda,
                                             int m0, int k0, int lane) {
    int m  = m0 + (lane & 15);
    int kh = (lane >> 4) << 3;                    // 0 or 8
    const bf16* p = A + (size_t)m * lda + (size_t)(k0 + kh);
    v8bf lo = *reinterpret_cast<const v8bf*>(p);
    v8bf hi = *reinterpret_cast<const v8bf*>(p + 16);
    return __builtin_shufflevector(lo, hi, 0, 1, 2, 3, 4, 5, 6, 7,
                                           8, 9, 10, 11, 12, 13, 14, 15);
}

// B-matrix 32x16 fragment where B[k][n] = W[n0+n][k0+k] (W row-major, ld=ldw).
__device__ __forceinline__ v16bf load_frag_bt(const bf16* __restrict__ W, int ldw,
                                              int n0, int k0, int lane) {
    int n  = n0 + (lane & 15);
    int kh = (lane >> 4) << 3;                    // 0 or 8
    const bf16* p = W + (size_t)n * ldw + (size_t)(k0 + kh);
    v8bf lo = *reinterpret_cast<const v8bf*>(p);
    v8bf hi = *reinterpret_cast<const v8bf*>(p + 16);
    return __builtin_shufflevector(lo, hi, 0, 1, 2, 3, 4, 5, 6, 7,
                                           8, 9, 10, 11, 12, 13, 14, 15);
}

__device__ __forceinline__ v8f wmma_bf16(v16bf a, v16bf b, v8f c) {
    return __builtin_amdgcn_wmma_f32_16x16x32_bf16(false, a, false, b,
                                                   (short)0, c, false, false);
}

// ---- 1) fp32 -> bf16 conversion -------------------------------------------

__global__ __launch_bounds__(256) void cvt_f32_bf16(const float* __restrict__ in,
                                                    bf16* __restrict__ out, int n) {
    int i = blockIdx.x * 256 + threadIdx.x;
    if (i < n) out[i] = (bf16)in[i];
}

// ---- 2) QKV GEMM: qkv = Xbf @ Wqkv^T, scattered into Q / K / Vt -----------
// wave = 32(M) x 64(N) register tile, double-buffered fragments.

__global__ __launch_bounds__(256) void qkv_gemm(const bf16* __restrict__ X,
                                                const bf16* __restrict__ Wqkv,
                                                bf16* __restrict__ Q,
                                                bf16* __restrict__ Kd,
                                                bf16* __restrict__ Vt) {
    int lane = threadIdx.x & 31;
    int gw   = (blockIdx.x * 256 + threadIdx.x) >> 5;
    const int nstrips = TDIM / 64;                // 48
    int tm = gw / nstrips;                        // 0..255
    int tn = gw % nstrips;
    int m0 = tm * 32, n0 = tn * 64;

    v8f acc[2][4] = {};
    v16bf a0 = load_frag_a(X, EMBED, m0,      0, lane);
    v16bf a1 = load_frag_a(X, EMBED, m0 + 16, 0, lane);
    v16bf b[4];
    #pragma unroll
    for (int j = 0; j < 4; ++j) b[j] = load_frag_bt(Wqkv, EMBED, n0 + j * 16, 0, lane);

    for (int k0 = 0; k0 < EMBED; k0 += 32) {
        int kn = (k0 + 32 < EMBED) ? k0 + 32 : 0;   // branchless prefetch index
        v16bf a0n = load_frag_a(X, EMBED, m0,      kn, lane);
        v16bf a1n = load_frag_a(X, EMBED, m0 + 16, kn, lane);
        v16bf bn[4];
        #pragma unroll
        for (int j = 0; j < 4; ++j) bn[j] = load_frag_bt(Wqkv, EMBED, n0 + j * 16, kn, lane);

        #pragma unroll
        for (int j = 0; j < 4; ++j) acc[0][j] = wmma_bf16(a0, b[j], acc[0][j]);
        #pragma unroll
        for (int j = 0; j < 4; ++j) acc[1][j] = wmma_bf16(a1, b[j], acc[1][j]);

        a0 = a0n; a1 = a1n;
        #pragma unroll
        for (int j = 0; j < 4; ++j) b[j] = bn[j];
    }

    // scatter: lane holds (row = i*16 + r + 8*kh, col = lane&15)
    int col = lane & 15, kh = lane >> 4;
    #pragma unroll
    for (int j = 0; j < 4; ++j) {
        int ng     = n0 + j * 16 + col;           // 0..3071
        int which  = ng >> 10;                    // 0:q 1:k 2:v
        int within = ng & 1023;
        int h = within >> 6, d = within & 63;
        #pragma unroll
        for (int i = 0; i < 2; ++i) {
            #pragma unroll
            for (int r = 0; r < 8; ++r) {
                int mg = m0 + i * 16 + r + (kh << 3);
                int bb = mg >> 11;                // / SEQ
                int n  = mg & 2047;
                bf16 val = (bf16)acc[i][j][r];
                if (which == 0)
                    Q[(((size_t)bb * NHEAD + h) * SEQ + n) * HDIM + d] = val;
                else if (which == 1)
                    Kd[(((size_t)bb * NHEAD + h) * SEQ + n) * HDIM + d] = val;
                else
                    Vt[(((size_t)bb * NHEAD + h) * HDIM + d) * SEQ + n] = val;
            }
        }
    }
}

// ---- 3) flash attention ----------------------------------------------------
// wave = one (b,h) + 16-query tile; 32-key tiles, online softmax,
// V-frags + next-K-frags prefetched ahead of the softmax VALU chunk.

__global__ __launch_bounds__(256) void attn_kernel(const bf16* __restrict__ Q,
                                                   const bf16* __restrict__ Km,
                                                   const bf16* __restrict__ Vt,
                                                   bf16* __restrict__ attnout) {
    __shared__ __align__(32) bf16 plds[8][16][32];   // per-wave P relayout buffer

    int lane = threadIdx.x & 31;
    int wib  = threadIdx.x >> 5;
    int gw   = blockIdx.x * 8 + wib;
    int qt   = gw & 127;                          // SEQ/16 = 128 query tiles
    int bh   = gw >> 7;                           // 0..63 = b*NHEAD + h
    int q0   = qt * 16;
    int col  = lane & 15, kh = lane >> 4;

    const bf16* Qp = Q  + (size_t)bh * SEQ * HDIM;
    const bf16* Kp = Km + (size_t)bh * SEQ * HDIM;
    const bf16* Vp = Vt + (size_t)bh * HDIM * SEQ;
    const float scale = 0.125f;                   // 64^-0.5

    float mrow[8], lrow[8];
    v8f o[4] = {v8f{}, v8f{}, v8f{}, v8f{}};
    #pragma unroll
    for (int r = 0; r < 8; ++r) { mrow[r] = -3.0e38f; lrow[r] = 0.0f; }

    v16bf qa0 = load_frag_a(Qp, HDIM, q0, 0, lane);
    v16bf qa1 = load_frag_a(Qp, HDIM, q0, 32, lane);

    // preload first K-tile fragments
    v16bf kb0 = load_frag_bt(Kp, HDIM, 0,  0,  lane);
    v16bf kb1 = load_frag_bt(Kp, HDIM, 0,  32, lane);
    v16bf kb2 = load_frag_bt(Kp, HDIM, 16, 0,  lane);
    v16bf kb3 = load_frag_bt(Kp, HDIM, 16, 32, lane);

    for (int jt = 0; jt < SEQ; jt += 32) {
        // in-flight early: V frags for this tile + K frags for next tile
        v16bf vb[4];
        #pragma unroll
        for (int t4 = 0; t4 < 4; ++t4)
            vb[t4] = load_frag_bt(Vp, SEQ, t4 * 16, jt, lane);
        int jtn = (jt + 32 < SEQ) ? jt + 32 : 0;  // branchless prefetch index
        v16bf kb0n = load_frag_bt(Kp, HDIM, jtn,      0,  lane);
        v16bf kb1n = load_frag_bt(Kp, HDIM, jtn,      32, lane);
        v16bf kb2n = load_frag_bt(Kp, HDIM, jtn + 16, 0,  lane);
        v16bf kb3n = load_frag_bt(Kp, HDIM, jtn + 16, 32, lane);

        // S = (Q @ K^T): two 16x16 C tiles (cols jt..+15, jt+16..+31)
        v8f s0 = v8f{}, s1 = v8f{};
        s0 = wmma_bf16(qa0, kb0, s0);
        s0 = wmma_bf16(qa1, kb1, s0);
        s1 = wmma_bf16(qa0, kb2, s1);
        s1 = wmma_bf16(qa1, kb3, s1);

        // online softmax over the 32 new columns (rows live in 16-lane halves)
        float p0[8], p1[8];
        #pragma unroll
        for (int r = 0; r < 8; ++r) {
            float v0 = s0[r] * scale, v1 = s1[r] * scale;
            float t = fmaxf(v0, v1);
            #pragma unroll
            for (int off = 1; off < 16; off <<= 1)
                t = fmaxf(t, __shfl_xor(t, off, 32));
            float mnew  = fmaxf(mrow[r], t);
            float alpha = __expf(mrow[r] - mnew);
            p0[r] = __expf(v0 - mnew);
            p1[r] = __expf(v1 - mnew);
            float ts = p0[r] + p1[r];
            #pragma unroll
            for (int off = 1; off < 16; off <<= 1)
                ts += __shfl_xor(ts, off, 32);
            lrow[r] = lrow[r] * alpha + ts;
            mrow[r] = mnew;
            #pragma unroll
            for (int t4 = 0; t4 < 4; ++t4) o[t4][r] = o[t4][r] * alpha;
        }

        // P (C-layout) -> LDS -> A-fragment layout
        #pragma unroll
        for (int r = 0; r < 8; ++r) {
            plds[wib][r + (kh << 3)][col]      = (bf16)p0[r];
            plds[wib][r + (kh << 3)][16 + col] = (bf16)p1[r];
        }
        const bf16* pp = &plds[wib][lane & 15][kh << 3];
        v8bf lo = *reinterpret_cast<const v8bf*>(pp);
        v8bf hi = *reinterpret_cast<const v8bf*>(pp + 16);
        v16bf pa = __builtin_shufflevector(lo, hi, 0, 1, 2, 3, 4, 5, 6, 7,
                                                   8, 9, 10, 11, 12, 13, 14, 15);

        // O += P @ V : B[j][d] = Vt[d][jt + j] (prefetched above)
        #pragma unroll
        for (int t4 = 0; t4 < 4; ++t4)
            o[t4] = wmma_bf16(pa, vb[t4], o[t4]);

        kb0 = kb0n; kb1 = kb1n; kb2 = kb2n; kb3 = kb3n;
    }

    // normalize and write attnout[b][n][h*64 + d]  (bf16, [B,N,D])
    int bb = bh >> 4, h = bh & 15;
    #pragma unroll
    for (int t4 = 0; t4 < 4; ++t4) {
        #pragma unroll
        for (int r = 0; r < 8; ++r) {
            int n = q0 + r + (kh << 3);
            float v = o[t4][r] / lrow[r];
            attnout[((size_t)bb * SEQ + n) * EMBED + h * HDIM + t4 * 16 + col] = (bf16)v;
        }
    }
}

// ---- 4) out-projection: out = attnout @ Wout^T + b_out (fp32 out) ----------
// wave = 32x64 register tile, double-buffered fragments.

__global__ __launch_bounds__(256) void proj_gemm(const bf16* __restrict__ A,
                                                 const bf16* __restrict__ Wout,
                                                 const float* __restrict__ bias,
                                                 float* __restrict__ out) {
    int lane = threadIdx.x & 31;
    int gw   = (blockIdx.x * 256 + threadIdx.x) >> 5;
    const int nstrips = EMBED / 64;               // 16
    int tm = gw / nstrips;                        // 0..255
    int tn = gw % nstrips;
    int m0 = tm * 32, n0 = tn * 64;

    v8f acc[2][4] = {};
    v16bf a0 = load_frag_a(A, EMBED, m0,      0, lane);
    v16bf a1 = load_frag_a(A, EMBED, m0 + 16, 0, lane);
    v16bf b[4];
    #pragma unroll
    for (int j = 0; j < 4; ++j) b[j] = load_frag_bt(Wout, EMBED, n0 + j * 16, 0, lane);

    for (int k0 = 0; k0 < EMBED; k0 += 32) {
        int kn = (k0 + 32 < EMBED) ? k0 + 32 : 0;
        v16bf a0n = load_frag_a(A, EMBED, m0,      kn, lane);
        v16bf a1n = load_frag_a(A, EMBED, m0 + 16, kn, lane);
        v16bf bn[4];
        #pragma unroll
        for (int j = 0; j < 4; ++j) bn[j] = load_frag_bt(Wout, EMBED, n0 + j * 16, kn, lane);

        #pragma unroll
        for (int j = 0; j < 4; ++j) acc[0][j] = wmma_bf16(a0, b[j], acc[0][j]);
        #pragma unroll
        for (int j = 0; j < 4; ++j) acc[1][j] = wmma_bf16(a1, b[j], acc[1][j]);

        a0 = a0n; a1 = a1n;
        #pragma unroll
        for (int j = 0; j < 4; ++j) b[j] = bn[j];
    }

    int col = lane & 15, kh = lane >> 4;
    #pragma unroll
    for (int j = 0; j < 4; ++j) {
        int n = n0 + j * 16 + col;
        float bv = bias[n];
        #pragma unroll
        for (int i = 0; i < 2; ++i) {
            #pragma unroll
            for (int r = 0; r < 8; ++r) {
                int mg = m0 + i * 16 + r + (kh << 3);
                out[(size_t)mg * EMBED + n] = acc[i][j][r] + bv;
            }
        }
    }
}

// ---------------------------------------------------------------------------

extern "C" void kernel_launch(void* const* d_in, const int* in_sizes, int n_in,
                              void* d_out, int out_size, void* d_ws, size_t ws_size,
                              hipStream_t stream) {
    const float* x     = (const float*)d_in[0];   // [4,2048,1024]
    const float* w_qkv = (const float*)d_in[1];   // [3072,1024]
    const float* w_out = (const float*)d_in[2];   // [1024,1024]
    const float* b_out = (const float*)d_in[3];   // [1024]
    float* out = (float*)d_out;                   // [4,2048,1024] fp32

    const size_t NX    = (size_t)MROWS * EMBED;       // 8388608
    const size_t NWQKV = (size_t)TDIM * EMBED;        // 3145728
    const size_t NWOUT = (size_t)EMBED * EMBED;       // 1048576
    const size_t NHEADBUF = (size_t)BATCH * NHEAD * SEQ * HDIM; // 8388608

    char* ws = (char*)d_ws;
    size_t off = 0;
    auto carve = [&](size_t elems) {
        bf16* p = (bf16*)(ws + off);
        off = (off + elems * sizeof(bf16) + 255) & ~(size_t)255;
        return p;
    };
    bf16* Xbf   = carve(NX);
    bf16* Wqbf  = carve(NWQKV);
    bf16* Wobf  = carve(NWOUT);
    bf16* Qb    = carve(NHEADBUF);   // [B,H,N,64]
    bf16* Kb    = carve(NHEADBUF);   // [B,H,N,64]
    bf16* Vtb   = carve(NHEADBUF);   // [B,H,64,N]
    bf16* AOb   = carve(NX);         // [B,N,D]
    (void)ws_size; (void)in_sizes; (void)n_in; (void)out_size;

    // 1) convert to bf16
    cvt_f32_bf16<<<(int)((NX    + 255) / 256), 256, 0, stream>>>(x,     Xbf,  (int)NX);
    cvt_f32_bf16<<<(int)((NWQKV + 255) / 256), 256, 0, stream>>>(w_qkv, Wqbf, (int)NWQKV);
    cvt_f32_bf16<<<(int)((NWOUT + 255) / 256), 256, 0, stream>>>(w_out, Wobf, (int)NWOUT);

    // 2) QKV GEMM: (8192/32)*(3072/64) = 12288 waves, 8 waves/block
    qkv_gemm<<<12288 / 8, 256, 0, stream>>>(Xbf, Wqbf, Qb, Kb, Vtb);

    // 3) attention: B*H*(N/16) = 8192 waves
    attn_kernel<<<8192 / 8, 256, 0, stream>>>(Qb, Kb, Vtb, AOb);

    // 4) out projection: (8192/32)*(1024/64) = 4096 waves
    proj_gemm<<<4096 / 8, 256, 0, stream>>>(AOb, Wobf, b_out, out);
}
